// DeformConvFCNSubNet_51092930953235
// MI455X (gfx1250) — matrix operand order repaired and checked by reading, not compiled, not run on hardware
//
#include <hip/hip_runtime.h>
#include <hip/hip_bf16.h>
#include <math.h>

// ---------------------------------------------------------------------------
// B=2, C=256, H=W=96, 4 layers of 3x3 deformable conv (dil=1).
// Main conv per layer == GEMM: M = B*H*W = 18432, N = 256, K = C*9 = 2304.
// Reduction order k = tap*256 + c  (tap fixed within each 32-wide K-chunk)
// so bilinear geometry is computed once per (pixel, tap) and reused across
// all 256 channels. TILE_N = 256 => every sample gathered exactly once.
// ---------------------------------------------------------------------------
#define BATCH   2
#define CH      256
#define IH      96
#define IW      96
#define HWSZ    (IH * IW)          // 9216
#define TAPS    9
#define OFFCH   18
#define LAYERS  4

#define TILE_M  64                 // pixels per workgroup
#define KC      32                 // K-chunk (channels; tap fixed per chunk)
#define NCHUNK  (TAPS * (CH / KC)) // 72
#define APAD    40                 // LDS row stride in bf16 (32 + 8 pad)

typedef __attribute__((ext_vector_type(16))) __bf16 v16bf;
typedef __attribute__((ext_vector_type(8)))  __bf16 v8bf;
typedef __attribute__((ext_vector_type(8)))  float  v8f;

union Frag { v16bf v; v8bf h[2]; };

static __device__ __forceinline__ int iclamp(int v, int lo, int hi) {
    return v < lo ? lo : (v > hi ? hi : v);
}

// ---------------------------------------------------------------------------
// Offset conv: 256 -> 18 channels, 3x3 SAME, zero pad. Direct conv (7% FLOPs).
// ---------------------------------------------------------------------------
__global__ __launch_bounds__(256)
void off_conv_kernel(const float* __restrict__ x,
                     const float* __restrict__ ow,   // (18, 256, 3, 3)
                     const float* __restrict__ ob,   // (18,)
                     float* __restrict__ off)        // (B, 18, H, W)
{
    const int t  = blockIdx.x * 256 + threadIdx.x;   // exact: B*18*HW threads
    const int xx = t % IW;
    const int y  = (t / IW) % IH;
    const int ch = (t / HWSZ) % OFFCH;
    const int b  = t / (HWSZ * OFFCH);

    const float* xb  = x + (size_t)b * CH * HWSZ;
    const float* wch = ow + (size_t)ch * CH * TAPS;

    float acc = ob[ch];
    for (int c = 0; c < CH; ++c) {
        const float* xc = xb + c * HWSZ;
        const float* wc = wch + c * TAPS;
#pragma unroll
        for (int ky = 0; ky < 3; ++ky) {
            const int yy = y + ky - 1;
            if (yy < 0 || yy >= IH) continue;
#pragma unroll
            for (int kx = 0; kx < 3; ++kx) {
                const int xr = xx + kx - 1;
                if (xr < 0 || xr >= IW) continue;
                acc = fmaf(xc[yy * IW + xr], wc[ky * 3 + kx], acc);
            }
        }
    }
    off[t] = acc;
}

// ---------------------------------------------------------------------------
// Repack dw (O=256, C=256, 3, 3) f32 into WMMA B-fragment layout, bf16:
//   bpk[((tap*8 + kb)*16 + nt)*512 + lane*16 + h]
// where lane<16 holds column n = nt*16+lane, K = h (0..15);
//       lane>=16 holds n = nt*16+lane-16,  K = 16+h.
// GEMM waves then load B fragments as two contiguous 16B global loads/lane.
// ---------------------------------------------------------------------------
__global__ __launch_bounds__(256)
void repack_w_kernel(const float* __restrict__ dw, __bf16* __restrict__ bpk)
{
    const int idx  = blockIdx.x * 256 + threadIdx.x;  // 9*8*16*512 = 589824
    const int h    = idx & 15;
    const int lane = (idx >> 4) & 31;
    const int nt   = (idx >> 9) & 15;
    const int kb   = (idx >> 13) & 7;
    const int tap  = idx >> 16;                       // 0..8

    const int n = nt * 16 + (lane & 15);
    const int c = kb * 32 + (lane >> 4) * 16 + h;
    bpk[idx] = (__bf16)dw[((size_t)n * CH + c) * TAPS + tap];
}

// ---------------------------------------------------------------------------
// Fused deformable-sample + GEMM (bf16 WMMA, f32 acc) + bias + ReLU.
// grid = (288): blockIdx.x = M-tile of 64 pixels (144 per image x 2 images).
// 256 threads = 8 waves: wm = wave&3 (16-row M tile), wn = wave>>2 (N half).
// Each wave: 1 A-frag x 8 B-frags = 8 v_wmma per K-chunk; 72 chunks.
// A tile is double-buffered in LDS: next chunk's gathers overlap WMMA.
// ---------------------------------------------------------------------------
__global__ __launch_bounds__(256)
void deform_gemm_kernel(const float* __restrict__ xin,  // (B, C, H, W)
                        const float* __restrict__ off,  // (B, 18, H, W)
                        const __bf16* __restrict__ bpk, // repacked weights
                        const float* __restrict__ db,   // (O,)
                        float* __restrict__ out)        // (B, O, H, W)
{
    __shared__ int    g_idx[TAPS][TILE_M][4];   // clamped y*W+x per corner
    __shared__ float  g_w  [TAPS][TILE_M][4];   // bilinear weight * validity
    __shared__ __bf16 Ald[2][TILE_M][APAD];     // double-buffered A tile

    const int mtile   = blockIdx.x;              // 0..287
    const int b       = mtile / (HWSZ / TILE_M); // /144
    const int pixbase = (mtile % (HWSZ / TILE_M)) * TILE_M;
    const int tid     = threadIdx.x;

    const float* xb   = xin + (size_t)b * CH * HWSZ;
    const float* offb = off + (size_t)b * OFFCH * HWSZ;

    // -------- geometry prologue: 64 pixels x 9 taps ----------------------
    for (int it = tid; it < TILE_M * TAPS; it += 256) {
        const int p   = it % TILE_M;
        const int tap = it / TILE_M;
        const int pix = pixbase + p;
        const int yy  = pix / IW, xx = pix % IW;
        const int ky  = tap / 3,  kx = tap % 3;
        const float oy = offb[(tap * 2 + 0) * HWSZ + pix];
        const float ox = offb[(tap * 2 + 1) * HWSZ + pix];
        const float py = (float)(yy - 1 + ky) + oy;
        const float px = (float)(xx - 1 + kx) + ox;
        const float y0f = floorf(py), x0f = floorf(px);
        const float wy1 = py - y0f, wy0 = 1.0f - wy1;
        const float wx1 = px - x0f, wx0 = 1.0f - wx1;
        const int y0 = (int)y0f, x0 = (int)x0f;
#pragma unroll
        for (int q = 0; q < 4; ++q) {
            const int dy = q >> 1, dx = q & 1;
            const int yi = y0 + dy, xi = x0 + dx;
            const bool v = (yi >= 0) & (yi < IH) & (xi >= 0) & (xi < IW);
            g_idx[tap][p][q] = iclamp(yi, 0, IH - 1) * IW + iclamp(xi, 0, IW - 1);
            const float wq = (dy ? wy1 : wy0) * (dx ? wx1 : wx0);
            g_w[tap][p][q] = v ? wq : 0.0f;
        }
    }
    __syncthreads();

    const int wave = tid >> 5;
    const int lane = tid & 31;
    const int wm   = wave & 3;          // 16-row M tile within 64
    const int wn   = wave >> 2;         // N half (8 x 16 columns)

    v8f acc[8] = {};

    // A-fill mapping: pixel = tid&63, 8 contiguous channels per thread
    // => one packed ds_store_b128 per chunk per thread.
    const int pfill = tid & (TILE_M - 1);
    const int cfb   = (tid >> 6) * 8;

    auto fillA = [&](int ci, int buf) {
        const int tap = ci / (CH / KC);
        const int c0  = (ci % (CH / KC)) * KC;
        const int*   gi = g_idx[tap][pfill];
        const float* gw = g_w[tap][pfill];
        const int   i0 = gi[0], i1 = gi[1], i2 = gi[2], i3 = gi[3];
        const float w0 = gw[0], w1 = gw[1], w2 = gw[2], w3 = gw[3];
        v8bf av = {};
#pragma unroll
        for (int u = 0; u < 8; ++u) {
            const float* xc = xb + (size_t)(c0 + cfb + u) * HWSZ;
            const float s = xc[i0] * w0 + xc[i1] * w1 +
                            xc[i2] * w2 + xc[i3] * w3;
            av[u] = (__bf16)s;
        }
        *(v8bf*)&Ald[buf][pfill][cfb] = av;
    };

    fillA(0, 0);
    __syncthreads();

    const int lidx = lane & 15;
    const int kcA  = (lane >> 4) * 16;  // A frag: K 0-15 vs 16-31
    const int arow = wm * 16 + lidx;

    for (int ci = 0; ci < NCHUNK; ++ci) {
        const int buf = ci & 1;
        const int tap = ci / (CH / KC);
        const int kb  = ci % (CH / KC);

        // A fragment from LDS (current buffer)
        Frag af;
        af.h[0] = *(const v8bf*)&Ald[buf][arow][kcA];
        af.h[1] = *(const v8bf*)&Ald[buf][arow][kcA + 8];

        // overlap: gather-fill next chunk into the other buffer
        if (ci + 1 < NCHUNK) fillA(ci + 1, buf ^ 1);

        // B fragments straight from repacked global (L2-resident) + WMMA
        const __bf16* btap = bpk + ((size_t)(tap * 8 + kb) * 16) * 512 + lane * 16;
#pragma unroll
        for (int j = 0; j < 8; ++j) {
            const int nt = wn * 8 + j;
            const __bf16* bp = btap + (size_t)nt * 512;
            Frag bfr;
            bfr.h[0] = *(const v8bf*)(bp);
            bfr.h[1] = *(const v8bf*)(bp + 8);
            acc[j] = __builtin_amdgcn_wmma_f32_16x16x32_bf16(
                false, af.v, false, bfr.v, (short)0, acc[j], false, false);
        }
        __syncthreads();   // fill(buf^1) visible; all reads of buf done
    }

    // -------- epilogue: bias + ReLU, contiguous 8-float runs per lane ----
    float* outb = out + (size_t)b * CH * HWSZ;
    const int nn = lane & 15;
    const int mh = (lane >> 4) * 8;     // C/D layout: lanes 16-31 hold M = r+8
#pragma unroll
    for (int j = 0; j < 8; ++j) {
        const int o = (wn * 8 + j) * 16 + nn;
        const float bias = db[o];
        float* op = outb + (size_t)o * HWSZ + pixbase + wm * 16 + mh;
#pragma unroll
        for (int r = 0; r < 8; ++r) {
            const float v = acc[j][r] + bias;
            op[r] = v > 0.0f ? v : 0.0f;
        }
    }
}

// ---------------------------------------------------------------------------
// Host-side chaining: L0: x -> ws_x ; L1: ws_x -> out ; L2: out -> ws_x ;
// L3: ws_x -> out. Workspace: activations 18.9MB + offsets 1.33MB +
// repacked weights 1.13MB (bf16)  ~= 21.4 MB.
// ---------------------------------------------------------------------------
extern "C" void kernel_launch(void* const* d_in, const int* in_sizes, int n_in,
                              void* d_out, int out_size, void* d_ws, size_t ws_size,
                              hipStream_t stream)
{
    const float* x     = (const float*)d_in[0];   // (2,256,96,96)
    const float* off_w = (const float*)d_in[1];   // (4,18,256,3,3)
    const float* off_b = (const float*)d_in[2];   // (4,18)
    const float* dw    = (const float*)d_in[3];   // (4,256,256,3,3)
    const float* db    = (const float*)d_in[4];   // (4,256)
    float* outp = (float*)d_out;

    float*  ws_x   = (float*)d_ws;                        // BATCH*CH*HWSZ f32
    float*  ws_off = ws_x + (size_t)BATCH * CH * HWSZ;    // BATCH*18*HWSZ f32
    __bf16* ws_bpk = (__bf16*)(ws_off + (size_t)BATCH * OFFCH * HWSZ);

    const dim3 offGrid((BATCH * OFFCH * HWSZ) / 256);
    const dim3 rpkGrid((TAPS * (CH / KC) * 16 * 512) / 256);  // 2304
    const dim3 gemmGrid((BATCH * HWSZ) / TILE_M);             // 288

    for (int l = 0; l < LAYERS; ++l) {
        const float* in  = (l == 0) ? x : ((l & 1) ? ws_x : outp);
        float*       dst = (l & 1) ? outp : ws_x;

        const float* owl = off_w + (size_t)l * OFFCH * CH * TAPS;
        const float* obl = off_b + (size_t)l * OFFCH;
        const float* dwl = dw    + (size_t)l * CH * CH * TAPS;
        const float* dbl = db    + (size_t)l * CH;

        off_conv_kernel<<<offGrid, 256, 0, stream>>>(in, owl, obl, ws_off);
        repack_w_kernel<<<rpkGrid, 256, 0, stream>>>(dwl, ws_bpk);
        deform_gemm_kernel<<<gemmGrid, 256, 0, stream>>>(in, ws_off, ws_bpk,
                                                         dbl, dst);
    }
}
